// MultiHeadAttention_84301618086184
// MI455X (gfx1250) — matrix operand order repaired and checked by reading, not compiled, not run on hardware
//
#include <hip/hip_runtime.h>

typedef __attribute__((ext_vector_type(16))) __bf16 v16bf;
typedef __attribute__((ext_vector_type(8)))  __bf16 v8bf;
typedef __attribute__((ext_vector_type(8)))  float  v8f;
typedef __attribute__((ext_vector_type(4)))  int    v4i;

#define NB   4
#define NS   2048
#define NH   16
#define HD   64
#define DM   1024          // D_IN == D_OUT
#define ROWS (NB * NS)     // 8192

#if __has_builtin(__builtin_amdgcn_global_load_async_to_lds_b128) && \
    __has_builtin(__builtin_amdgcn_s_wait_asynccnt)
#define HAVE_ASYNC_LDS 1
#else
#define HAVE_ASYNC_LDS 0
#endif

__device__ __forceinline__ v16bf cat16(v8bf a, v8bf b) {
  v16bf r;
#pragma unroll
  for (int i = 0; i < 8; ++i) { r[i] = a[i]; r[i + 8] = b[i]; }
  return r;
}

__device__ __forceinline__ v8f wmma_bf16(v16bf a, v16bf b, v8f c) {
  // D = A(16x32 bf16) * B(32x16 bf16) + C(16x16 f32)
  return __builtin_amdgcn_wmma_f32_16x16x32_bf16(false, a, false, b, (short)0, c,
                                                 false, false);
}

// 16-byte global -> LDS copy via async DMA (falls back to a normal copy).
__device__ __forceinline__ void cp16(const __bf16* g, __bf16* l) {
#if HAVE_ASYNC_LDS
  __builtin_amdgcn_global_load_async_to_lds_b128(
      (__attribute__((address_space(1))) v4i*)(unsigned long long)(uintptr_t)g,
      (__attribute__((address_space(3))) v4i*)(unsigned)(uintptr_t)l,
      0, 0);
#else
  *(v8bf*)l = *(const v8bf*)g;
#endif
}

// Drain all outstanding async copies.
__device__ __forceinline__ void cp16_wait_all() {
#if HAVE_ASYNC_LDS
  __builtin_amdgcn_s_wait_asynccnt(0);
#endif
}

// Drain all but the newest batch of 3 (async loads complete in order, so this
// guarantees the previous tile's data has landed while the next tile's DMA
// stays in flight underneath the WMMAs).
__device__ __forceinline__ void cp16_wait_keep3() {
#if HAVE_ASYNC_LDS
  __builtin_amdgcn_s_wait_asynccnt(3);
#endif
}

// ---------------------------------------------------------------------------
// Prep A: xb = bf16(x)   (8M elements, 8 per thread)
// ---------------------------------------------------------------------------
__global__ __launch_bounds__(256) void cvt_x_kernel(const float* __restrict__ x,
                                                    __bf16* __restrict__ xb) {
  size_t i = ((size_t)blockIdx.x * 256 + threadIdx.x) * 8;
  float4 f0 = *(const float4*)(x + i);
  float4 f1 = *(const float4*)(x + i + 4);
  v8bf o;
  o[0] = (__bf16)f0.x; o[1] = (__bf16)f0.y; o[2] = (__bf16)f0.z; o[3] = (__bf16)f0.w;
  o[4] = (__bf16)f1.x; o[5] = (__bf16)f1.y; o[6] = (__bf16)f1.z; o[7] = (__bf16)f1.w;
  *(v8bf*)(xb + i) = o;
}

// ---------------------------------------------------------------------------
// Prep B: Wt[n][k] = bf16(W[k][n])  (LDS-staged 64x64 tile transpose)
// grid.z selects Wq/Wk/Wv/Wo
// ---------------------------------------------------------------------------
__global__ __launch_bounds__(256) void wt_kernel(
    const float* __restrict__ Wq, const float* __restrict__ Wk,
    const float* __restrict__ Wv, const float* __restrict__ Wo,
    __bf16* __restrict__ tq, __bf16* __restrict__ tk,
    __bf16* __restrict__ tv, __bf16* __restrict__ to) {
  __shared__ float T[64][65];
  const float* W =
      (blockIdx.z == 0) ? Wq : (blockIdx.z == 1 ? Wk : (blockIdx.z == 2 ? Wv : Wo));
  __bf16* Wt =
      (blockIdx.z == 0) ? tq : (blockIdx.z == 1 ? tk : (blockIdx.z == 2 ? tv : to));
  const int tid = threadIdx.x;
  const int n0 = blockIdx.x * 64, k0 = blockIdx.y * 64;
#pragma unroll
  for (int i = 0; i < 4; ++i) {
    int c = tid + i * 256;               // 1024 float4 chunks
    int row = c >> 4, c4 = c & 15;
    const float4 f = *(const float4*)&W[(size_t)(k0 + row) * DM + n0 + c4 * 4];
    T[row][c4 * 4 + 0] = f.x; T[row][c4 * 4 + 1] = f.y;
    T[row][c4 * 4 + 2] = f.z; T[row][c4 * 4 + 3] = f.w;
  }
  __syncthreads();
#pragma unroll
  for (int i = 0; i < 2; ++i) {
    int c = tid + i * 256;               // 512 chunks of 8 halves
    int nr = c >> 3, kc = (c & 7) * 8;
    v8bf o;
#pragma unroll
    for (int j = 0; j < 8; ++j) o[j] = (__bf16)T[kc + j][nr];
    *(v8bf*)&Wt[(size_t)(n0 + nr) * DM + k0 + kc] = o;
  }
}

// ---------------------------------------------------------------------------
// Kernel 1: fused QKV projection (bf16, double-buffered async tile staging).
// C = xb[8192x1024] @ W; out bf16 [B,H,S,HD]; grid.z selects head matrix.
// ---------------------------------------------------------------------------
__global__ __launch_bounds__(256) void qkv_gemm_kernel(
    const __bf16* __restrict__ xb,
    const __bf16* __restrict__ Wtq, const __bf16* __restrict__ Wtk,
    const __bf16* __restrict__ Wtv,
    __bf16* __restrict__ qo, __bf16* __restrict__ ko, __bf16* __restrict__ vo) {
  constexpr int BM = 128, BN = 64, BK = 32, LDA = BK + 8, LDB = BK + 8;
  __shared__ __bf16 As[2][BM * LDA];   // [m][k]
  __shared__ __bf16 Bs[2][BN * LDB];   // [n][k] (weights pre-transposed)

  const __bf16* W   = (blockIdx.z == 0) ? Wtq : (blockIdx.z == 1 ? Wtk : Wtv);
  __bf16*       dst = (blockIdx.z == 0) ? qo  : (blockIdx.z == 1 ? ko  : vo);

  const int tid = threadIdx.x;
  const int lane = tid & 31, wave = tid >> 5;
  const int wr = wave & 3, wc = wave >> 2;          // 4x2 wave grid, 32x32 each
  const int m = lane & 15, hi = lane >> 4;
  const int m0 = blockIdx.y * BM, n0 = blockIdx.x * BN;

  // per-thread chunk coordinates (3 x 16B chunks per tile)
  const int ar0 = tid >> 2,         ao0 = (tid & 3) * 8;          // A chunk 0
  const int ar1 = (tid + 256) >> 2, ao1 = ((tid + 256) & 3) * 8;  // A chunk 1
  const int br  = tid >> 2,         bo_ = (tid & 3) * 8;          // B chunk

  auto issue = [&](int kk, int buf) {
    cp16(xb + (size_t)(m0 + ar0) * DM + kk + ao0, &As[buf][ar0 * LDA + ao0]);
    cp16(xb + (size_t)(m0 + ar1) * DM + kk + ao1, &As[buf][ar1 * LDA + ao1]);
    cp16(W  + (size_t)(n0 + br)  * DM + kk + bo_, &Bs[buf][br  * LDB + bo_]);
  };

  v8f acc[2][2] = {};
  constexpr int NT = DM / BK;  // 32 k-tiles
  issue(0, 0);

  for (int t = 0; t < NT; ++t) {
    const int buf = t & 1;
    if (t + 1 < NT) {
      issue((t + 1) * BK, buf ^ 1);  // next tile's DMA runs under this tile's WMMAs
      cp16_wait_keep3();             // in-order completion => tile t has landed
    } else {
      cp16_wait_all();
    }
    __syncthreads();

#pragma unroll
    for (int mi = 0; mi < 2; ++mi) {
      // A frag: lanes 0-15 hold K {0-7,16-23}, lanes 16-31 K {8-15,24-31}
      const __bf16* ap = &As[buf][(wr * 32 + mi * 16 + m) * LDA + hi * 8];
      v16bf a = cat16(*(const v8bf*)ap, *(const v8bf*)(ap + 16));
#pragma unroll
      for (int ni = 0; ni < 2; ++ni) {
        // B frag: lanes 0-15 hold K {0-7,16-23}, lanes 16-31 K {8-15,24-31}
        const __bf16* bp = &Bs[buf][(wc * 32 + ni * 16 + m) * LDB + hi * 8];
        v16bf b = cat16(*(const v8bf*)bp, *(const v8bf*)(bp + 16));
        acc[mi][ni] = wmma_bf16(a, b, acc[mi][ni]);
      }
    }
    __syncthreads();  // everyone done reading buf before it is overwritten
  }

#pragma unroll
  for (int mi = 0; mi < 2; ++mi)
#pragma unroll
    for (int ni = 0; ni < 2; ++ni)
#pragma unroll
      for (int r = 0; r < 8; ++r) {
        int rg = m0 + wr * 32 + mi * 16 + hi * 8 + r;
        int cg = n0 + wc * 32 + ni * 16 + m;
        int b = rg >> 11, s = rg & (NS - 1);
        int h = cg >> 6, d = cg & (HD - 1);
        dst[(((size_t)(b * NH + h)) * NS + s) * HD + d] = (__bf16)acc[mi][ni][r];
      }
}

// ---------------------------------------------------------------------------
// Kernel 2: causal flash attention.  Block = 128 query rows of one (b,h);
// wave = 16 rows.  64-key tiles, online softmax, K tile staged async.
// ---------------------------------------------------------------------------
__global__ __launch_bounds__(256) void attn_kernel(
    const __bf16* __restrict__ q, const __bf16* __restrict__ k,
    const __bf16* __restrict__ v, __bf16* __restrict__ ctx) {
  constexpr int BQ = 128, BKV = 64, LDK = HD + 8, LDV = BKV + 8, LDP = HD + 8;
  __shared__ __bf16 Ks[BKV * LDK];       // [key][d]
  __shared__ __bf16 Vs[HD * LDV];        // transposed [d][key]
  __shared__ __bf16 Ps[8 * 16 * LDP];    // per-wave P scratch [wave][row][key]

  const int tid = threadIdx.x, lane = tid & 31, wave = tid >> 5;
  const int m = lane & 15, hi = lane >> 4;
  const int bh = blockIdx.y;
  const int q0 = blockIdx.x * BQ;
  const size_t base = (size_t)bh * NS * HD;

  v16bf qa[2];
  {
    const __bf16* qp = q + base + (size_t)(q0 + wave * 16 + m) * HD;
#pragma unroll
    for (int c = 0; c < 2; ++c) {
      const __bf16* p = qp + c * 32 + hi * 8;
      qa[c] = cat16(*(const v8bf*)p, *(const v8bf*)(p + 16));
    }
  }

  float mrow[8], lrow[8];
#pragma unroll
  for (int r = 0; r < 8; ++r) { mrow[r] = -1e30f; lrow[r] = 0.f; }
  v8f o[4] = {};

  const int jmax = q0 + BQ;
  for (int j0 = 0; j0 < jmax; j0 += BKV) {
    // K tile: verbatim 16B chunks -> async DMA into LDS
#pragma unroll
    for (int i = 0; i < 2; ++i) {
      int c = tid + i * 256;               // 512 chunks of 8 halves
      int row = c >> 3, off = (c & 7) * 8;
      cp16(k + base + (size_t)(j0 + row) * HD + off, &Ks[row * LDK + off]);
      // V tile needs [d][key] transpose: regular load + scattered ds stores
      v8bf vv = *(const v8bf*)(v + base + (size_t)(j0 + row) * HD + off);
#pragma unroll
      for (int jj = 0; jj < 8; ++jj) Vs[(off + jj) * LDV + row] = vv[jj];
    }
    cp16_wait_all();
    __syncthreads();

    // scores: S(16x64) = Q(16x64) . K^T
    v8f sc[4];
#pragma unroll
    for (int nt = 0; nt < 4; ++nt) {
      v8f z = {};
#pragma unroll
      for (int c = 0; c < 2; ++c) {
        const __bf16* bp = &Ks[(nt * 16 + m) * LDK + c * 32 + hi * 8];
        v16bf bfr = cat16(*(const v8bf*)bp, *(const v8bf*)(bp + 16));
        z = wmma_bf16(qa[c], bfr, z);
      }
      sc[nt] = z;
    }

    // causal mask + 1/sqrt(64) scale + online softmax
#pragma unroll
    for (int r = 0; r < 8; ++r) {
      int qg = q0 + wave * 16 + hi * 8 + r;
      float vmax = -1e30f;
#pragma unroll
      for (int nt = 0; nt < 4; ++nt) {
        int kg = j0 + nt * 16 + m;
        float sv = sc[nt][r] * 0.125f;
        if (kg > qg) sv = -1e30f;
        sc[nt][r] = sv;
        vmax = fmaxf(vmax, sv);
      }
      vmax = fmaxf(vmax, __shfl_xor(vmax, 1, 32));
      vmax = fmaxf(vmax, __shfl_xor(vmax, 2, 32));
      vmax = fmaxf(vmax, __shfl_xor(vmax, 4, 32));
      vmax = fmaxf(vmax, __shfl_xor(vmax, 8, 32));
      float mnew = fmaxf(mrow[r], vmax);
      float alpha = __expf(mrow[r] - mnew);
      float rsum = 0.f;
#pragma unroll
      for (int nt = 0; nt < 4; ++nt) {
        float p = __expf(sc[nt][r] - mnew);
        rsum += p;
        Ps[(wave * 16 + hi * 8 + r) * LDP + nt * 16 + m] = (__bf16)p;
      }
      rsum += __shfl_xor(rsum, 1, 32);
      rsum += __shfl_xor(rsum, 2, 32);
      rsum += __shfl_xor(rsum, 4, 32);
      rsum += __shfl_xor(rsum, 8, 32);
      lrow[r] = lrow[r] * alpha + rsum;
      mrow[r] = mnew;
#pragma unroll
      for (int nt = 0; nt < 4; ++nt) o[nt][r] *= alpha;
    }

    // O += P(16x64) . V(64x64)
    v16bf pa[2];
#pragma unroll
    for (int c = 0; c < 2; ++c) {
      const __bf16* pp = &Ps[(wave * 16 + m) * LDP + c * 32 + hi * 8];
      pa[c] = cat16(*(const v8bf*)pp, *(const v8bf*)(pp + 16));
    }
#pragma unroll
    for (int nt = 0; nt < 4; ++nt)
#pragma unroll
      for (int c = 0; c < 2; ++c) {
        const __bf16* bp = &Vs[(nt * 16 + m) * LDV + c * 32 + hi * 8];
        v16bf bfr = cat16(*(const v8bf*)bp, *(const v8bf*)(bp + 16));
        o[nt] = wmma_bf16(pa[c], bfr, o[nt]);
      }
    __syncthreads();
  }

  // normalize, write ctx as [B, S, H*HD] bf16
  const int b = bh >> 4, h = bh & (NH - 1);
#pragma unroll
  for (int r = 0; r < 8; ++r) {
    float inv = 1.f / lrow[r];
    int srow = q0 + wave * 16 + hi * 8 + r;
    size_t rowb = ((size_t)(b * NS + srow)) * DM + h * HD;
#pragma unroll
    for (int nt = 0; nt < 4; ++nt)
      ctx[rowb + nt * 16 + m] = (__bf16)(o[nt][r] * inv);
  }
}

// ---------------------------------------------------------------------------
// Kernel 3: output projection (double-buffered async).
// out f32 = ctx_bf16 @ Wo + bo (Wo pre-transposed)
// ---------------------------------------------------------------------------
__global__ __launch_bounds__(256) void out_proj_kernel(
    const __bf16* __restrict__ ctx, const __bf16* __restrict__ Wto,
    const float* __restrict__ bo, float* __restrict__ out) {
  constexpr int BM = 128, BN = 64, BK = 32, LDA = BK + 8, LDB = BK + 8;
  __shared__ __bf16 As[2][BM * LDA];
  __shared__ __bf16 Bs[2][BN * LDB];

  const int tid = threadIdx.x, lane = tid & 31, wave = tid >> 5;
  const int wr = wave & 3, wc = wave >> 2;
  const int m = lane & 15, hi = lane >> 4;
  const int m0 = blockIdx.y * BM, n0 = blockIdx.x * BN;

  const int ar0 = tid >> 2,         ao0 = (tid & 3) * 8;
  const int ar1 = (tid + 256) >> 2, ao1 = ((tid + 256) & 3) * 8;
  const int br  = tid >> 2,         bo_ = (tid & 3) * 8;

  auto issue = [&](int kk, int buf) {
    cp16(ctx + (size_t)(m0 + ar0) * DM + kk + ao0, &As[buf][ar0 * LDA + ao0]);
    cp16(ctx + (size_t)(m0 + ar1) * DM + kk + ao1, &As[buf][ar1 * LDA + ao1]);
    cp16(Wto + (size_t)(n0 + br)  * DM + kk + bo_, &Bs[buf][br  * LDB + bo_]);
  };

  v8f acc[2][2] = {};
  constexpr int NT = DM / BK;
  issue(0, 0);

  for (int t = 0; t < NT; ++t) {
    const int buf = t & 1;
    if (t + 1 < NT) {
      issue((t + 1) * BK, buf ^ 1);
      cp16_wait_keep3();
    } else {
      cp16_wait_all();
    }
    __syncthreads();

#pragma unroll
    for (int mi = 0; mi < 2; ++mi) {
      const __bf16* ap = &As[buf][(wr * 32 + mi * 16 + m) * LDA + hi * 8];
      v16bf a = cat16(*(const v8bf*)ap, *(const v8bf*)(ap + 16));
#pragma unroll
      for (int ni = 0; ni < 2; ++ni) {
        const __bf16* bp = &Bs[buf][(wc * 32 + ni * 16 + m) * LDB + hi * 8];
        v16bf b = cat16(*(const v8bf*)bp, *(const v8bf*)(bp + 16));
        acc[mi][ni] = wmma_bf16(a, b, acc[mi][ni]);
      }
    }
    __syncthreads();
  }

#pragma unroll
  for (int mi = 0; mi < 2; ++mi)
#pragma unroll
    for (int ni = 0; ni < 2; ++ni) {
      int cg = n0 + wc * 32 + ni * 16 + m;
      float bias = bo[cg];
#pragma unroll
      for (int r = 0; r < 8; ++r) {
        int rg = m0 + wr * 32 + mi * 16 + hi * 8 + r;
        out[(size_t)rg * DM + cg] = acc[mi][ni][r] + bias;
      }
    }
}

// ---------------------------------------------------------------------------
extern "C" void kernel_launch(void* const* d_in, const int* in_sizes, int n_in,
                              void* d_out, int out_size, void* d_ws,
                              size_t ws_size, hipStream_t stream) {
  const float* x  = (const float*)d_in[0];
  const float* Wq = (const float*)d_in[1];
  const float* Wk = (const float*)d_in[2];
  const float* Wv = (const float*)d_in[3];
  const float* Wo = (const float*)d_in[4];
  const float* bo = (const float*)d_in[5];
  float* out = (float*)d_out;

  const size_t elems = (size_t)ROWS * DM;   // 8M
  const size_t welems = (size_t)DM * DM;    // 1M
  __bf16* xb  = (__bf16*)d_ws;              // bf16 x
  __bf16* wtq = xb + elems;                 // transposed bf16 weights
  __bf16* wtk = wtq + welems;
  __bf16* wtv = wtk + welems;
  __bf16* wto = wtv + welems;
  __bf16* qb  = wto + welems;               // [B,H,S,HD]
  __bf16* kb  = qb + elems;
  __bf16* vb  = kb + elems;
  __bf16* cb  = vb + elems;                 // [B,S,H*HD]

  cvt_x_kernel<<<dim3((unsigned)(elems / 2048)), 256, 0, stream>>>(x, xb);
  wt_kernel<<<dim3(DM / 64, DM / 64, 4), 256, 0, stream>>>(
      Wq, Wk, Wv, Wo, wtq, wtk, wtv, wto);
  qkv_gemm_kernel<<<dim3(DM / 64, ROWS / 128, 3), 256, 0, stream>>>(
      xb, wtq, wtk, wtv, qb, kb, vb);
  attn_kernel<<<dim3(NS / 128, NB * NH), 256, 0, stream>>>(qb, kb, vb, cb);
  out_proj_kernel<<<dim3(DM / 64, ROWS / 128), 256, 0, stream>>>(cb, wto, bo, out);
}
